// CharRNN_13993003450943
// MI455X (gfx1250) — matrix operand order repaired and checked by reading, not compiled, not run on hardware
//
#include <hip/hip_runtime.h>

typedef __attribute__((ext_vector_type(16))) __bf16 v16bf;
typedef __attribute__((ext_vector_type(8)))  __bf16 v8bf;
typedef __attribute__((ext_vector_type(8)))  float  v8f;

#define B_ 256
#define L_ 512
#define V_ 32
#define E_ 64
#define H_ 256
#define TB 16        // batch rows per block (WMMA M)
#define HSTRIDE 264  // padded bf16 row stride for h in LDS (bank-conflict free)

union AFrag { v16bf v; v8bf h[2]; };

__device__ __forceinline__ float fast_tanh(float x) {
#if __has_builtin(__builtin_amdgcn_tanhf)
  return __builtin_amdgcn_tanhf(x);
#elif __has_builtin(__builtin_amdgcn_exp2f) && __has_builtin(__builtin_amdgcn_rcpf)
  // tanh(x) = 1 - 2/(exp2(2*log2e*x)+1)
  float e = __builtin_amdgcn_exp2f(x * 2.8853900817779268f);
  return 1.0f - 2.0f * __builtin_amdgcn_rcpf(e + 1.0f);
#else
  return tanhf(x);
#endif
}

__launch_bounds__(512, 1)
__global__ void rnn_fused_kernel(const int* __restrict__ x, const float* __restrict__ h0,
                                 const float* __restrict__ emb, const float* __restrict__ Wih,
                                 const float* __restrict__ Whh, const float* __restrict__ bh,
                                 const float* __restrict__ Wout, const float* __restrict__ bout,
                                 float* __restrict__ out)
{
  __shared__ __bf16 xwtab[V_ * H_];                         // 16 KB: emb@W_ih + b_h (bf16)
  __shared__ __align__(16) __bf16 hbuf[2][TB][HSTRIDE];     // 16.9 KB: double-buffered h
  __shared__ __align__(8) unsigned char xloc[L_ * TB];      // 8 KB: tokens, transposed
  __shared__ __align__(16) __bf16 woutfrag[2 * 8 * 32 * 16];// 16 KB: W_out pre-swizzled frags

  const int tid    = threadIdx.x;
  const int lane   = tid & 31;
  const int w      = tid >> 5;        // wave id 0..15
  const int ln     = lane & 15;       // N within tile
  const int hi     = (lane >> 4) & 1; // half-wave selector
  const int batch0 = blockIdx.x * TB;
  const int n0     = w * 16;          // this wave's H-column tile

  // ---- xwtab[v][n] = b_h[n] + sum_e emb[v][e] * W_ih[e][n]  (replaces input-proj GEMM)
  for (int i = tid; i < V_ * H_; i += 512) {
    int v = i >> 8, n = i & (H_ - 1);
    float s = bh[n];
    #pragma unroll 8
    for (int e = 0; e < E_; ++e) s += emb[v * E_ + e] * Wih[e * H_ + n];
    xwtab[i] = (__bf16)s;
  }
  // ---- tokens for this block's batch rows, transposed to [t][m]
  for (int i = tid; i < TB * L_; i += 512) {
    int m = i >> 9, t = i & (L_ - 1);
    xloc[t * TB + m] = (unsigned char)x[(batch0 + m) * L_ + t];
  }
  // ---- h0 -> hbuf[0] (bf16)
  for (int i = tid; i < TB * H_; i += 512) {
    int m = i >> 8, c = i & (H_ - 1);
    hbuf[0][m][c] = (__bf16)h0[(batch0 + m) * H_ + c];
  }
  // ---- W_out fragments, pre-swizzled per consuming (wave, kc, lane): 16 contiguous bf16
  //      element e for lane l (v-tile wv): Wout[(kc*32 + ((l&16)?16:0) + e)*V + wv*16 + (l&15)]
  for (int i = tid; i < 2 * 8 * 32 * 16; i += 512) {
    int e     = i & 15;
    int lane_ = (i >> 4) & 31;
    int kc    = (i >> 9) & 7;
    int wv    = i >> 12;
    int k = kc * 32 + ((lane_ & 16) ? 16 : 0) + e;
    woutfrag[i] = (__bf16)Wout[k * V_ + wv * 16 + (lane_ & 15)];
  }

  // ---- W_hh B-fragments: registers for the whole kernel (the only resident weight set).
  // B frags in contiguous k order used here (same as A): lane l holds col n0+ln; lanes 0-15: K=0..15, lanes 16-31: K=16..31.
  v16bf Bhh[8];
  #pragma unroll
  for (int kc = 0; kc < 8; ++kc) {
    v16bf f;
    #pragma unroll
    for (int e = 0; e < 16; ++e) {
      int k = kc * 32 + hi * 16 + e;
      f[e] = (__bf16)Whh[k * H_ + n0 + ln];
    }
    Bhh[kc] = f;
  }
  // Pin: force one-time bf16 materialization; forbid remat/sinking of cvts into the loop.
  #pragma unroll
  for (int kc = 0; kc < 8; ++kc) asm volatile("" : "+v"(Bhh[kc]));

  float boutv = (w < 2) ? bout[n0 + ln] : 0.f;

  __syncthreads();

  const long hfin_off = (long)B_ * L_ * V_;
  const __bf16* mywout = &woutfrag[(w * 8) * 32 * 16 + lane * 16]; // valid for w<2

  for (int t = 0; t < L_; ++t) {
    const int rp = t & 1, wp = rp ^ 1;      // double buffer: read h_{t-1}, write h_t
    v8f acc  = {};
    v8f accL = {};
    const bool doL = (w < 2) && (t > 0);    // wave-uniform: logits for step t-1, reusing A frags

    #pragma unroll
    for (int kc = 0; kc < 8; ++kc) {
      // A frag in contiguous k order used here (same as B): lane l = row ln; one contiguous 16-elem K run at
      // kc*32 + hi*16  -> two aligned ds_load_b128.
      AFrag a;
      const __bf16* base = &hbuf[rp][ln][kc * 32 + hi * 16];
      a.h[0] = *(const v8bf*)(base);
      a.h[1] = *(const v8bf*)(base + 8);
      acc = __builtin_amdgcn_wmma_f32_16x16x32_bf16(false, a.v, false, Bhh[kc],
                                                    (short)0, acc, false, false);
      if (doL) {
        AFrag b;
        const __bf16* wb = mywout + kc * 32 * 16;
        b.h[0] = *(const v8bf*)(wb);
        b.h[1] = *(const v8bf*)(wb + 8);
        accL = __builtin_amdgcn_wmma_f32_16x16x32_bf16(false, a.v, false, b.v,
                                                       (short)0, accL, false, false);
      }
    }

    // pipelined logits store for timestep t-1
    if (doL) {
      #pragma unroll
      for (int j = 0; j < 8; ++j) {
        int m = j + hi * 8;
        out[((long)(batch0 + m) * L_ + (t - 1)) * V_ + n0 + ln] = accL[j] + boutv;
      }
    }

    // 8 tokens for this half-wave's rows: single ds_load_b64
    unsigned long long tk = *(const unsigned long long*)&xloc[t * TB + hi * 8];

    #pragma unroll
    for (int j = 0; j < 8; ++j) {
      int m   = j + hi * 8;
      int tok = (int)((tk >> (8 * j)) & 0xff);
      float val = acc[j] + (float)xwtab[tok * H_ + n0 + ln];
      float h = fast_tanh(val);
      hbuf[wp][m][n0 + ln] = (__bf16)h;
      if (t == L_ - 1)
        out[hfin_off + (long)(batch0 + m) * H_ + n0 + ln] = h;   // h_final (f32)
    }
    __syncthreads();
  }

  // logits for the final timestep (h_{L-1} sits in hbuf[0] since L is even)
  if (w < 2) {
    v8f accL = {};
    #pragma unroll
    for (int kc = 0; kc < 8; ++kc) {
      AFrag a, b;
      const __bf16* base = &hbuf[0][ln][kc * 32 + hi * 16];
      a.h[0] = *(const v8bf*)(base);
      a.h[1] = *(const v8bf*)(base + 8);
      const __bf16* wb = mywout + kc * 32 * 16;
      b.h[0] = *(const v8bf*)(wb);
      b.h[1] = *(const v8bf*)(wb + 8);
      accL = __builtin_amdgcn_wmma_f32_16x16x32_bf16(false, a.v, false, b.v,
                                                     (short)0, accL, false, false);
    }
    #pragma unroll
    for (int j = 0; j < 8; ++j) {
      int m = j + hi * 8;
      out[((long)(batch0 + m) * L_ + (L_ - 1)) * V_ + n0 + ln] = accL[j] + boutv;
    }
  }
}

extern "C" void kernel_launch(void* const* d_in, const int* in_sizes, int n_in,
                              void* d_out, int out_size, void* d_ws, size_t ws_size,
                              hipStream_t stream) {
  const int*   x    = (const int*)d_in[0];
  const float* h0   = (const float*)d_in[1];
  const float* emb  = (const float*)d_in[2];
  const float* Wih  = (const float*)d_in[3];
  const float* Whh  = (const float*)d_in[4];
  const float* bh   = (const float*)d_in[5];
  const float* Wout = (const float*)d_in[6];
  const float* bout = (const float*)d_in[7];
  float* out = (float*)d_out;

  dim3 grid(B_ / TB);   // 16 blocks, one per 16-row batch tile
  dim3 block(512);      // 16 wave32 waves
  hipLaunchKernelGGL(rnn_fused_kernel, grid, block, 0, stream,
                     x, h0, emb, Wih, Whh, bh, Wout, bout, out);
}
